// MoELayer_60833916781078
// MI455X (gfx1250) — matrix operand order repaired and checked by reading, not compiled, not run on hardware
//
#include <hip/hip_runtime.h>
#include <hip/hip_bf16.h>

// ---------------------------------------------------------------------------
// MoE layer for MI455X (gfx1250): routed top-2 grouped GEMM via bf16 WMMA.
// Round 4: block tile 128x128 (8 waves x 16x128 each, 8 WMMA accumulators)
// -> 33% less L2 traffic per FLOP, A fragment amortized over 8 WMMAs.
// ---------------------------------------------------------------------------

#define NTOK  16384
#define DD    1024
#define HH    2048
#define EE    8

#define TM 128       // tokens per block tile
#define TN 128       // H columns per block tile
#define TK 32        // K (D) step == WMMA bf16 K
#define TKP (TK + 8) // padded LDS row: 40 bf16 = 80 B = 20 dwords (bank spread)

typedef __attribute__((ext_vector_type(16))) __bf16 v16bf;
typedef __attribute__((ext_vector_type(8)))  __bf16 v8bf;
typedef __attribute__((ext_vector_type(2)))  __bf16 v2bf;
typedef __attribute__((ext_vector_type(8)))  float  v8f;

#if __has_builtin(__builtin_amdgcn_sched_barrier)
#define SCHED_FENCE() __builtin_amdgcn_sched_barrier(0)
#else
#define SCHED_FENCE()
#endif

// workspace layout (bytes)
#define WS_COUNTS 0                         // unsigned counts[8] (cursors)
#define WS_ENT    64                        // float entPart[1024]
#define WS_LIST   8192                      // int   tokList[E][NTOK]
#define WS_PROB   (WS_LIST + EE*NTOK*4)     // float probList[E][NTOK]

// ---------------------------------------------------------------------------
// Kernel 1: gating. One wave per token (wave32).
// ---------------------------------------------------------------------------
__global__ __launch_bounds__(512) void moe_gate_kernel(
    const float* __restrict__ x, const float* __restrict__ gw,
    const float* __restrict__ gb, unsigned* __restrict__ counts,
    float* __restrict__ entPart, int* __restrict__ tokList,
    float* __restrict__ probList)
{
    __shared__ float sEnt[16];
    const int lane  = threadIdx.x & 31;
    const int wave  = threadIdx.x >> 5;
    const int token = blockIdx.x * 16 + wave;

    float acc[EE];
#pragma unroll
    for (int e = 0; e < EE; ++e) acc[e] = 0.0f;

    const float* xr = x + (size_t)token * DD;
    for (int d = lane; d < DD; d += 32) {               // coalesced over lanes
        const float  xv = xr[d];
        const float4* g4 = (const float4*)(gw + (size_t)d * EE);
        const float4 g0 = g4[0], g1 = g4[1];
        acc[0] += xv * g0.x; acc[1] += xv * g0.y;
        acc[2] += xv * g0.z; acc[3] += xv * g0.w;
        acc[4] += xv * g1.x; acc[5] += xv * g1.y;
        acc[6] += xv * g1.z; acc[7] += xv * g1.w;
    }
#pragma unroll
    for (int e = 0; e < EE; ++e) {
#pragma unroll
        for (int off = 16; off > 0; off >>= 1)
            acc[e] += __shfl_xor(acc[e], off, 32);
    }

    float l[EE], p[EE];
    float mx = -1e30f;
#pragma unroll
    for (int e = 0; e < EE; ++e) { l[e] = acc[e] + gb[e]; mx = fmaxf(mx, l[e]); }
    float z = 0.0f;
#pragma unroll
    for (int e = 0; e < EE; ++e) { p[e] = expf(l[e] - mx); z += p[e]; }
    const float inv = 1.0f / z;
    float ent = 0.0f;
#pragma unroll
    for (int e = 0; e < EE; ++e) { p[e] *= inv; ent -= p[e] * logf(p[e] + 1e-8f); }

    // top-2 (stable: ties pick lower index, matching lax.top_k)
    int i0 = 0; float p0 = p[0];
#pragma unroll
    for (int e = 1; e < EE; ++e) if (p[e] > p0) { p0 = p[e]; i0 = e; }
    int i1 = -1; float p1 = -1.0f;
#pragma unroll
    for (int e = 0; e < EE; ++e) if (e != i0 && p[e] > p1) { p1 = p[e]; i1 = e; }

    if (lane == 0) {
        sEnt[wave] = ent;
        unsigned pos0 = atomicAdd(&counts[i0], 1u);
        tokList[i0 * NTOK + pos0]  = token;
        probList[i0 * NTOK + pos0] = p0;
        unsigned pos1 = atomicAdd(&counts[i1], 1u);
        tokList[i1 * NTOK + pos1]  = token;
        probList[i1 * NTOK + pos1] = p1;
    }
    __syncthreads();
    if (threadIdx.x == 0) {
        float s = 0.0f;
#pragma unroll
        for (int w = 0; w < 16; ++w) s += sEnt[w];      // fixed order
        entPart[blockIdx.x] = s;
    }
}

// ---------------------------------------------------------------------------
// Kernel 2: per-expert grouped GEMM with gather, bf16 WMMA, fp32 accum.
// Block 128(tok) x 128(H); 8 waves, each owns a 16x128 strip = 8 accumulators.
// Pipeline: store regs->LDS(p) | barrier | issue next globals | [fence]
// 8 WMMAs from LDS(p) | [fence].
// ---------------------------------------------------------------------------
__global__ __launch_bounds__(256) void moe_gemm_kernel(
    const float* __restrict__ x, const float* __restrict__ ew,
    const float* __restrict__ eb, const unsigned* __restrict__ counts,
    const int* __restrict__ tokList, const float* __restrict__ probList,
    float* __restrict__ out)
{
    const int e = blockIdx.z;
    const unsigned cnt = counts[e];
    const int m0 = blockIdx.y * TM;
    if ((unsigned)m0 >= cnt) return;
    const int h0 = blockIdx.x * TN;

    __shared__ __align__(64) __bf16 As[2][TM][TKP];   // 2 x 128 x 40 bf16 = 20 KB
    __shared__ __align__(64) __bf16 Bs[2][TN][TKP];   // 2 x 128 x 40 bf16 = 20 KB
    __shared__ int   sTok[TM];
    __shared__ float sProb[TM];

    const int tid   = threadIdx.x;
    const int lane  = tid & 31;
    const int wid   = tid >> 5;
    const int half  = lane >> 4;                      // 0 | 1 lane group
    const int l16   = lane & 15;
    const int mBase = wid * 16;                       // 8 M-waves

    if (tid < TM) {
        const int r = m0 + tid;
        if ((unsigned)r < cnt) {
            sTok[tid]  = tokList[(size_t)e * NTOK + r];
            sProb[tid] = probList[(size_t)e * NTOK + r];
        } else { sTok[tid] = -1; sProb[tid] = 0.0f; }
    }
    __syncthreads();

    // A staging: thread converts 16 fp32 of half a gathered row
    const int rowA  = tid >> 1;                       // 0..127
    const int csegA = (tid & 1) * 16;                 // 0 or 16
    const int tokA  = (sTok[rowA] < 0) ? 0 : sTok[rowA];
    const float* xa = x + (size_t)tokA * DD + csegA;

    // B staging: thread owns k-row-pairs (2*kp0, 2*kp0+1) and (2*kp1, 2*kp1+1)
    // at 4 consecutive h columns; global reads stay 512 B-contiguous per wave.
    const int kp0 = wid;                              // 0..7
    const int kp1 = wid + 8;                          // 8..15
    const int h4  = lane * 4;                         // 0..124
    const float* wb = ew + (size_t)e * DD * HH + h0 + h4;

    float4 ra0, ra1, ra2, ra3, rb0, rb1, rb2, rb3;
#define LOAD_TILE(K0)                                                       \
    ra0 = *(const float4*)(xa + (K0));                                      \
    ra1 = *(const float4*)(xa + (K0) + 4);                                  \
    ra2 = *(const float4*)(xa + (K0) + 8);                                  \
    ra3 = *(const float4*)(xa + (K0) + 12);                                 \
    rb0 = *(const float4*)(wb + (size_t)((K0) + 2*kp0    ) * HH);           \
    rb1 = *(const float4*)(wb + (size_t)((K0) + 2*kp0 + 1) * HH);           \
    rb2 = *(const float4*)(wb + (size_t)((K0) + 2*kp1    ) * HH);           \
    rb3 = *(const float4*)(wb + (size_t)((K0) + 2*kp1 + 1) * HH);

    const v8f zero = {0,0,0,0,0,0,0,0};
    v8f c[8] = { zero, zero, zero, zero, zero, zero, zero, zero };

    LOAD_TILE(0);
    int p = 0;
#pragma unroll 2
    for (int k0 = 0; k0 < DD; k0 += TK, p ^= 1) {
        // ---- store staged registers as bf16 into LDS buffer p ----
        {
            v8bf av0, av1;
            av0[0] = (__bf16)ra0.x; av0[1] = (__bf16)ra0.y;
            av0[2] = (__bf16)ra0.z; av0[3] = (__bf16)ra0.w;
            av0[4] = (__bf16)ra1.x; av0[5] = (__bf16)ra1.y;
            av0[6] = (__bf16)ra1.z; av0[7] = (__bf16)ra1.w;
            av1[0] = (__bf16)ra2.x; av1[1] = (__bf16)ra2.y;
            av1[2] = (__bf16)ra2.z; av1[3] = (__bf16)ra2.w;
            av1[4] = (__bf16)ra3.x; av1[5] = (__bf16)ra3.y;
            av1[6] = (__bf16)ra3.z; av1[7] = (__bf16)ra3.w;
            *(v8bf*)&As[p][rowA][csegA]     = av0;
            *(v8bf*)&As[p][rowA][csegA + 8] = av1;

            const float* f0 = (const float*)&rb0;
            const float* f1 = (const float*)&rb1;
            const float* f2 = (const float*)&rb2;
            const float* f3 = (const float*)&rb3;
#pragma unroll
            for (int j = 0; j < 4; ++j) {             // packed k-pair stores (b32)
                v2bf t0; t0[0] = (__bf16)f0[j]; t0[1] = (__bf16)f1[j];
                *(v2bf*)&Bs[p][h4 + j][2 * kp0] = t0;
                v2bf t1; t1[0] = (__bf16)f2[j]; t1[1] = (__bf16)f3[j];
                *(v2bf*)&Bs[p][h4 + j][2 * kp1] = t1;
            }
        }
        __syncthreads();

        // ---- issue next tile's globals AFTER the barrier; fence so their
        //      consumers (converts) cannot be hoisted above the WMMAs ----
        if (k0 + TK < DD) { LOAD_TILE(k0 + TK); }
        SCHED_FENCE();

        // ---- A fragment: lanes<16 K{0..7,16..23}, lanes>=16 K{8..15,24..31}
        const int mrow = mBase + l16;
        const v8bf* arow = (const v8bf*)&As[p][mrow][0];
        const v8bf alo = arow[half];
        const v8bf ahi = arow[2 + half];
        const v16bf a = __builtin_shufflevector(alo, ahi,
            0,1,2,3,4,5,6,7,8,9,10,11,12,13,14,15);

        // ---- 8 WMMAs: 16x16 tiles across the wave's 128 H columns ----
#pragma unroll
        for (int j = 0; j < 8; ++j) {
            const int hcol = j * 16 + l16;
            const v8bf* brow = (const v8bf*)&Bs[p][hcol][0];
            const v16bf b = __builtin_shufflevector(
                brow[half], brow[2 + half],
                0,1,2,3,4,5,6,7,8,9,10,11,12,13,14,15);
            c[j] = __builtin_amdgcn_wmma_f32_16x16x32_bf16(
                false, a, false, b, (short)0, c[j], false, false);
        }
        SCHED_FENCE();
    }
#undef LOAD_TILE

    // ---- epilogue: acc VGPR r -> token row mBase + 8*half + r ----
    int   tokR[8]; float pR[8];
#pragma unroll
    for (int r = 0; r < 8; ++r) {
        const int m = mBase + half * 8 + r;
        tokR[r] = sTok[m]; pR[r] = sProb[m];
    }
#pragma unroll
    for (int j = 0; j < 8; ++j) {
        const int hcol = h0 + j * 16 + l16;
        const float bias = eb[(size_t)e * HH + hcol];
#pragma unroll
        for (int r = 0; r < 8; ++r) {
            if (tokR[r] >= 0)   // exactly 2 atomic contributions per out element
                atomicAdd(out + (size_t)tokR[r] * HH + hcol,
                          pR[r] * (c[j][r] + bias));
        }
    }
}

// ---------------------------------------------------------------------------
// Kernel 3: deterministic fixed-order loss reduction.
// ---------------------------------------------------------------------------
__global__ __launch_bounds__(256) void moe_loss_kernel(
    const float* __restrict__ entPart, const unsigned* __restrict__ counts,
    float* __restrict__ lossOut)
{
    __shared__ float s[256];
    const int t = threadIdx.x;
    float v = entPart[t] + entPart[t + 256] + entPart[t + 512] + entPart[t + 768];
    s[t] = v;
    __syncthreads();
    for (int off = 128; off > 0; off >>= 1) {
        if (t < off) s[t] += s[t + off];
        __syncthreads();
    }
    if (t == 0) {
        float loss = 0.05f * (s[0] / (float)NTOK);      // ENTROPY_W * mean entropy
        for (int e = 0; e < EE; ++e) {
            const float u = (float)counts[e] / ((float)NTOK + 1e-8f);
            loss += fmaxf(u - 0.4f, 0.0f);              // overuse penalty
        }
        lossOut[0] = loss;
    }
}

// ---------------------------------------------------------------------------
extern "C" void kernel_launch(void* const* d_in, const int* in_sizes, int n_in,
                              void* d_out, int out_size, void* d_ws, size_t ws_size,
                              hipStream_t stream)
{
    (void)in_sizes; (void)n_in; (void)ws_size;
    const float* x  = (const float*)d_in[0];
    const float* gw = (const float*)d_in[1];
    const float* gb = (const float*)d_in[2];
    const float* ew = (const float*)d_in[3];
    const float* eb = (const float*)d_in[4];
    float* out = (float*)d_out;

    char* ws = (char*)d_ws;
    unsigned* counts   = (unsigned*)(ws + WS_COUNTS);
    float*    entPart  = (float*)(ws + WS_ENT);
    int*      tokList  = (int*)(ws + WS_LIST);
    float*    probList = (float*)(ws + WS_PROB);

    hipMemsetAsync(d_out, 0, (size_t)out_size * sizeof(float), stream);
    hipMemsetAsync(counts, 0, 64, stream);

    moe_gate_kernel<<<NTOK / 16, 512, 0, stream>>>(
        x, gw, gb, counts, entPart, tokList, probList);

    dim3 gg(HH / TN, NTOK / TM, EE);
    moe_gemm_kernel<<<gg, 256, 0, stream>>>(
        x, ew, eb, counts, tokList, probList, out);

    moe_loss_kernel<<<1, 256, 0, stream>>>(
        entPart, counts, out + (size_t)NTOK * HH);
}
